// MNDRE_84808424227322
// MI455X (gfx1250) — compile-verified, matrix operand order checked by reading, not run on hardware
//
#include <hip/hip_runtime.h>
#include <math.h>

// ---- problem dims ----
#define Bn   256
#define Tn   2049
#define Dn   8
#define Sn   128          // state
#define Hn   256          // hidden
#define WINn 16
#define OUTn 32
#define Nn   128          // windows

typedef __attribute__((ext_vector_type(16))) __bf16 v16bf;
typedef __attribute__((ext_vector_type(8)))  float  v8f;

// Fast activations: v_exp_f32 / v_log_f32 hardware transcendentals.
__device__ __forceinline__ float softplusf_(float x) {
  // stable softplus: max(x,0) + log(1 + exp(-|x|))
  return fmaxf(x, 0.f) + __logf(1.f + __expf(-fabsf(x)));
}
__device__ __forceinline__ float sigmoidf_(float x) {
  return 1.f / (1.f + __expf(-x));
}
__device__ __forceinline__ float tanhf_(float x) {
  float e = __expf(2.f * x);        // inf -> 1, 0 -> -1 : correct limits
  return 1.f - 2.f / (e + 1.f);
}

// ---- WMMA fragment helpers (16x16x32 bf16, f32 accum) ----
// A (16x32): lanes 0-15 row=lane, v0..3 K={0..7}, v4..7 K={16..23};
//            lanes 16-31 same row, K={8..15},{24..31}.  B mirrors with cols.
__device__ __forceinline__ v16bf ldfrag(const __bf16* rowp, int kbase) {
  const int khalf = ((threadIdx.x >> 4) & 1) << 3;
  union { v16bf v; unsigned u[8]; } f;
  const __bf16* p = rowp + kbase + khalf;
#pragma unroll
  for (int r = 0; r < 4; ++r) f.u[r]     = *(const unsigned*)(p + 2 * r);
#pragma unroll
  for (int r = 0; r < 4; ++r) f.u[4 + r] = *(const unsigned*)(p + 16 + 2 * r);
  return f.v;
}

__device__ __forceinline__ v8f wmma_bf16(v16bf a, v16bf b, v8f c) {
  return __builtin_amdgcn_wmma_f32_16x16x32_bf16(false, a, false, b, (short)0, c,
                                                 false, false);
}

// ---------------- kernel 0: f32 -> bf16 weight conversion ----------------
__global__ void cvt_bf16_kernel(const float* __restrict__ src,
                                __bf16* __restrict__ dst, int ncount) {
  int i = blockIdx.x * blockDim.x + threadIdx.x;
  if (i < ncount) dst[i] = (__bf16)src[i];
}

// ---------------- kernel 1: windowed depth-2 log-signatures ----------------
// per (b,n): lvl1[8] and antisymmetric C[8][8] = 0.5*(S - S^T) (Levy area)
__global__ void sig_kernel(const float* __restrict__ x,
                           float* __restrict__ lvl1ws,
                           float* __restrict__ Cws) {
  int tid = blockIdx.x * blockDim.x + threadIdx.x;
  if (tid >= Bn * Nn) return;
  int b = tid / Nn, n = tid % Nn;
  const float* xp = x + ((size_t)b * Tn + (size_t)n * WINn) * Dn;
  float l1[Dn];
  float S[Dn][Dn];
#pragma unroll
  for (int i = 0; i < Dn; ++i) {
    l1[i] = 0.f;
#pragma unroll
    for (int j = 0; j < Dn; ++j) S[i][j] = 0.f;
  }
  for (int w = 0; w < WINn; ++w) {
    float dx[Dn];
#pragma unroll
    for (int d = 0; d < Dn; ++d) dx[d] = xp[(w + 1) * Dn + d] - xp[w * Dn + d];
#pragma unroll
    for (int i = 0; i < Dn; ++i)
#pragma unroll
      for (int j = 0; j < Dn; ++j) S[i][j] += l1[i] * dx[j];  // pre (excl) x dx
#pragma unroll
    for (int d = 0; d < Dn; ++d) l1[d] += dx[d];
  }
  float* l1o = lvl1ws + (size_t)tid * Dn;
  float* Co  = Cws + (size_t)tid * 64;
#pragma unroll
  for (int i = 0; i < Dn; ++i) l1o[i] = l1[i];
#pragma unroll
  for (int i = 0; i < Dn; ++i)
#pragma unroll
    for (int j = 0; j < Dn; ++j) Co[i * 8 + j] = 0.5f * (S[i][j] - S[j][i]);
}

// ---------------- kernel 2: h0 = Wi1 @ softplus(Wi0 @ x0 + bi0) + bi1 -----
__global__ void __launch_bounds__(256)
init_kernel(const float* __restrict__ x, const float* __restrict__ Wi0,
            const float* __restrict__ bi0, const float* __restrict__ Wi1,
            const float* __restrict__ bi1, float* __restrict__ hidden) {
  __shared__ float a1[Hn];
  const int b = blockIdx.x;
  const int t = threadIdx.x;
  const float* x0 = x + (size_t)b * Tn * Dn;
  {
    float z = bi0[t];
#pragma unroll
    for (int d = 0; d < Dn; ++d) z += Wi0[t * Dn + d] * x0[d];
    a1[t] = softplusf_(z);
  }
  __syncthreads();
  if (t < Sn) {
    float z = bi1[t];
    for (int k = 0; k < Hn; ++k) z += Wi1[t * Hn + k] * a1[k];
    hidden[(size_t)b * (Nn + 1) * Sn + t] = z;
  }
}

// ---------------- kernel 3: one log-ODE step for all B ----------------
// LDS layout (bytes)
#define SM_HF32 0                       // 16*128*4  = 8192  h (f32)
#define SM_HACC (SM_HF32 + 8192)        // 8192            running hn
#define SM_L1   (SM_HACC + 8192)        // 512             lvl1 tile
#define SM_C    (SM_L1 + 512)           // 4096            C tile
#define SM_HB   (SM_C + 4096)           // 4096            h bf16
#define SM_A1   (SM_HB + 4096)          // 8192            H1 bf16
#define SM_A2   (SM_A1 + 8192)          // 8192            H2 bf16
#define SM_S1   (SM_A2 + 8192)          // 8192            sigmoid gate1 bf16
#define SM_S2   (SM_S1 + 8192)          // 8192            sigmoid gate2 bf16
#define SM_V    (SM_S2 + 8192)          // 32768           V bf16 (16x1024)
#define SM_US   (SM_V + 32768)          // 32768           seeded tangents
#define SM_T1   (SM_US + 32768)         // 65536
#define SM_T2   (SM_T1 + 65536)         // 65536
#define SM_TOT  (SM_T2 + 65536)         // 254464 (< 320KB WGP LDS)

__global__ void __launch_bounds__(256)
step_kernel(float* __restrict__ hidden, const __bf16* __restrict__ Wv0b,
            const __bf16* __restrict__ Wv1b, const __bf16* __restrict__ Wv2b,
            const float* __restrict__ bv0, const float* __restrict__ bv1,
            const float* __restrict__ bv2, const float* __restrict__ lvl1ws,
            const float* __restrict__ Cws, int n) {
  extern __shared__ char smem[];
  float*  hF32 = (float*)(smem + SM_HF32);
  float*  hAcc = (float*)(smem + SM_HACC);
  float*  sL1  = (float*)(smem + SM_L1);
  float*  sC   = (float*)(smem + SM_C);
  __bf16* hB   = (__bf16*)(smem + SM_HB);
  __bf16* A1   = (__bf16*)(smem + SM_A1);
  __bf16* A2   = (__bf16*)(smem + SM_A2);
  __bf16* S1   = (__bf16*)(smem + SM_S1);
  __bf16* S2   = (__bf16*)(smem + SM_S2);
  __bf16* Vt   = (__bf16*)(smem + SM_V);
  __bf16* Us   = (__bf16*)(smem + SM_US);
  __bf16* T1   = (__bf16*)(smem + SM_T1);
  __bf16* T2   = (__bf16*)(smem + SM_T2);

  const int tid  = threadIdx.x;
  const int lane = tid & 31;
  const int w    = tid >> 5;
  const int hi8  = ((lane >> 4) & 1) << 3;
  const int b0   = blockIdx.x * 16;

  // --- stage 0: load h tile + signature coefficients ---
  for (int e = tid; e < 16 * Sn; e += 256) {
    int bl = e >> 7, a = e & 127;
    float hv = hidden[((size_t)(b0 + bl) * (Nn + 1) + n) * Sn + a];
    hF32[e] = hv;
    hB[e]   = (__bf16)hv;
  }
  for (int e = tid; e < 16 * Dn; e += 256)
    sL1[e] = lvl1ws[((size_t)(b0 + (e >> 3)) * Nn + n) * Dn + (e & 7)];
  for (int e = tid; e < 16 * 64; e += 256)
    sC[e] = Cws[((size_t)(b0 + (e >> 6)) * Nn + n) * 64 + (e & 63)];
  __syncthreads();

  // --- stage 1: Z1 = h @ Wv0^T + bv0 -> H1, gate1  (16x256, K=128) ---
  {
    v16bf af[4];
    const __bf16* arow = hB + (lane & 15) * Sn;
#pragma unroll
    for (int kc = 0; kc < 4; ++kc) af[kc] = ldfrag(arow, kc * 32);
#pragma unroll
    for (int t = 0; t < 2; ++t) {
      int ct = w * 2 + t;
      int nc = ct * 16 + (lane & 15);
      const __bf16* brow = Wv0b + (size_t)nc * Sn;
      v8f acc;
#pragma unroll
      for (int r = 0; r < 8; ++r) acc[r] = bv0[nc];
#pragma unroll
      for (int kc = 0; kc < 4; ++kc)
        acc = wmma_bf16(af[kc], ldfrag(brow, kc * 32), acc);
#pragma unroll
      for (int r = 0; r < 8; ++r) {
        int m = r + hi8;
        float z = acc[r];
        A1[m * Hn + nc] = (__bf16)softplusf_(z);
        S1[m * Hn + nc] = (__bf16)sigmoidf_(z);
      }
    }
  }
  __syncthreads();

  // --- stage 2: Z2 = H1 @ Wv1^T + bv1 -> H2, gate2  (16x256, K=256) ---
  {
    v16bf af[8];
    const __bf16* arow = A1 + (lane & 15) * Hn;
#pragma unroll
    for (int kc = 0; kc < 8; ++kc) af[kc] = ldfrag(arow, kc * 32);
#pragma unroll
    for (int t = 0; t < 2; ++t) {
      int ct = w * 2 + t;
      int nc = ct * 16 + (lane & 15);
      const __bf16* brow = Wv1b + (size_t)nc * Hn;
      v8f acc;
#pragma unroll
      for (int r = 0; r < 8; ++r) acc[r] = bv1[nc];
#pragma unroll
      for (int kc = 0; kc < 8; ++kc)
        acc = wmma_bf16(af[kc], ldfrag(brow, kc * 32), acc);
#pragma unroll
      for (int r = 0; r < 8; ++r) {
        int m = r + hi8;
        float z = acc[r];
        A2[m * Hn + nc] = (__bf16)softplusf_(z);
        S2[m * Hn + nc] = (__bf16)sigmoidf_(z);
      }
    }
  }
  __syncthreads();

  // --- stage 3: V = tanh(H2 @ Wv2^T + bv2)  (16x1024, K=256) ---
  {
    v16bf af[8];
    const __bf16* arow = A2 + (lane & 15) * Hn;
#pragma unroll
    for (int kc = 0; kc < 8; ++kc) af[kc] = ldfrag(arow, kc * 32);
    for (int t = 0; t < 8; ++t) {
      int ct = w * 8 + t;
      int nc = ct * 16 + (lane & 15);
      const __bf16* brow = Wv2b + (size_t)nc * Hn;
      v8f acc;
#pragma unroll
      for (int r = 0; r < 8; ++r) acc[r] = bv2[nc];
#pragma unroll
      for (int kc = 0; kc < 8; ++kc)
        acc = wmma_bf16(af[kc], ldfrag(brow, kc * 32), acc);
#pragma unroll
      for (int r = 0; r < 8; ++r) {
        int m = r + hi8;
        Vt[m * 1024 + nc] = (__bf16)tanhf_(acc[r]);
      }
    }
  }
  __syncthreads();

  // --- stage 4: seed tangents with C folded in (linearity of the JVP):
  //     Us[(b,j),c] = sum_i C[b,i,j] * V[b,i*128+c]
  //     and start hAcc = h + sum_i lvl1_i * V_i  (level-1 term) ---
  {
    // 256 threads = 128 rows x 2 column-halves; paired u32 reads/writes
    int row = tid >> 1;
    int chalf = (tid & 1) * 64;
    int bl = row >> 3, j = row & 7;
    float cf[8];
#pragma unroll
    for (int i = 0; i < 8; ++i) cf[i] = sC[bl * 64 + i * 8 + j];
    const __bf16* vb = Vt + bl * 1024 + chalf;
    __bf16* uo = Us + row * Sn + chalf;
    for (int cc = 0; cc < 64; cc += 2) {
      float s0 = 0.f, s1 = 0.f;
#pragma unroll
      for (int i = 0; i < 8; ++i) {
        union { unsigned u; __bf16 h[2]; } uu;
        uu.u = *(const unsigned*)(vb + i * 128 + cc);
        s0 += cf[i] * (float)uu.h[0];
        s1 += cf[i] * (float)uu.h[1];
      }
      union { unsigned u; __bf16 h[2]; } wo;
      wo.h[0] = (__bf16)s0;
      wo.h[1] = (__bf16)s1;
      *(unsigned*)(uo + cc) = wo.u;
    }
  }
  {
    // 256 threads = 16 rows x 16 chunks of 8 states
    int bl = tid >> 4;
    int a0 = (tid & 15) * 8;
    float lf[8];
#pragma unroll
    for (int i = 0; i < 8; ++i) lf[i] = sL1[bl * 8 + i];
    const __bf16* vb = Vt + bl * 1024 + a0;
#pragma unroll
    for (int aa = 0; aa < 8; aa += 2) {
      float s0 = hF32[bl * Sn + a0 + aa];
      float s1 = hF32[bl * Sn + a0 + aa + 1];
#pragma unroll
      for (int i = 0; i < 8; ++i) {
        union { unsigned u; __bf16 h[2]; } uu;
        uu.u = *(const unsigned*)(vb + i * 128 + aa);
        s0 += lf[i] * (float)uu.h[0];
        s1 += lf[i] * (float)uu.h[1];
      }
      hAcc[bl * Sn + a0 + aa]     = s0;
      hAcc[bl * Sn + a0 + aa + 1] = s1;
    }
  }
  __syncthreads();

  // --- stage 5: T1 = gate1 .* (Us @ Wv0^T)   (128x256, K=128) ---
  {
    v16bf af[4];
    const __bf16* arow = Us + ((size_t)w * 16 + (lane & 15)) * Sn;
#pragma unroll
    for (int kc = 0; kc < 4; ++kc) af[kc] = ldfrag(arow, kc * 32);
    for (int ct = 0; ct < 16; ++ct) {
      int nc = ct * 16 + (lane & 15);
      const __bf16* brow = Wv0b + (size_t)nc * Sn;
      v8f acc;
#pragma unroll
      for (int r = 0; r < 8; ++r) acc[r] = 0.f;
#pragma unroll
      for (int kc = 0; kc < 4; ++kc)
        acc = wmma_bf16(af[kc], ldfrag(brow, kc * 32), acc);
#pragma unroll
      for (int r = 0; r < 8; ++r) {
        int row = w * 16 + r + hi8;
        int bl = row >> 3;
        T1[row * Hn + nc] = (__bf16)(acc[r] * (float)S1[bl * Hn + nc]);
      }
    }
  }
  __syncthreads();

  // --- stage 6: T2 = gate2 .* (T1 @ Wv1^T)   (128x256, K=256) ---
  {
    v16bf af[8];
    const __bf16* arow = T1 + ((size_t)w * 16 + (lane & 15)) * Hn;
#pragma unroll
    for (int kc = 0; kc < 8; ++kc) af[kc] = ldfrag(arow, kc * 32);
    for (int ct = 0; ct < 16; ++ct) {
      int nc = ct * 16 + (lane & 15);
      const __bf16* brow = Wv1b + (size_t)nc * Hn;
      v8f acc;
#pragma unroll
      for (int r = 0; r < 8; ++r) acc[r] = 0.f;
#pragma unroll
      for (int kc = 0; kc < 8; ++kc)
        acc = wmma_bf16(af[kc], ldfrag(brow, kc * 32), acc);
#pragma unroll
      for (int r = 0; r < 8; ++r) {
        int row = w * 16 + r + hi8;
        int bl = row >> 3;
        T2[row * Hn + nc] = (__bf16)(acc[r] * (float)S2[bl * Hn + nc]);
      }
    }
  }
  __syncthreads();

  // --- stage 7: per j: bracket delta = tanh' .* (T2[:,j] @ Wv2[j-block]^T);
  //     only block j of the output of tangent row j is needed.  wave w = j.
  {
    const int j = w;
    v16bf af[8];
    const __bf16* arow = T2 + ((size_t)(lane & 15) * 8 + j) * Hn;
#pragma unroll
    for (int kc = 0; kc < 8; ++kc) af[kc] = ldfrag(arow, kc * 32);
    for (int ct = 0; ct < 8; ++ct) {
      int ac = ct * 16 + (lane & 15);  // state index a in 0..127
      const __bf16* brow = Wv2b + ((size_t)j * Sn + ac) * Hn;
      v8f acc;
#pragma unroll
      for (int r = 0; r < 8; ++r) acc[r] = 0.f;
#pragma unroll
      for (int kc = 0; kc < 8; ++kc)
        acc = wmma_bf16(af[kc], ldfrag(brow, kc * 32), acc);
#pragma unroll
      for (int r = 0; r < 8; ++r) {
        int bl = r + hi8;
        float v = (float)Vt[bl * 1024 + j * 128 + ac];
        atomicAdd(&hAcc[bl * Sn + ac], (1.f - v * v) * acc[r]);
      }
    }
  }
  __syncthreads();

  // --- stage 8: write hn ---
  for (int e = tid; e < 16 * Sn; e += 256) {
    int bl = e >> 7, a = e & 127;
    hidden[((size_t)(b0 + bl) * (Nn + 1) + (n + 1)) * Sn + a] = hAcc[e];
  }
}

// ---------------- kernel 4: readout out = hidden @ Wr^T + br ----------------
__global__ void readout_kernel(const float* __restrict__ hidden,
                               const float* __restrict__ Wr,
                               const float* __restrict__ br,
                               float* __restrict__ out) {
  int tid = blockIdx.x * blockDim.x + threadIdx.x;
  const int total = Bn * (Nn + 1) * OUTn;
  if (tid >= total) return;
  int o = tid & (OUTn - 1);
  int bt = tid / OUTn;
  const float* hp = hidden + (size_t)bt * Sn;
  const float* wp = Wr + o * Sn;
  float z = br[o];
#pragma unroll 8
  for (int a = 0; a < Sn; ++a) z += hp[a] * wp[a];
  out[tid] = z;
}

// ---------------- host launcher ----------------
extern "C" void kernel_launch(void* const* d_in, const int* in_sizes, int n_in,
                              void* d_out, int out_size, void* d_ws,
                              size_t ws_size, hipStream_t stream) {
  (void)in_sizes; (void)n_in; (void)out_size; (void)ws_size;
  const float* x   = (const float*)d_in[1];
  const float* Wi0 = (const float*)d_in[2];
  const float* bi0 = (const float*)d_in[3];
  const float* Wi1 = (const float*)d_in[4];
  const float* bi1 = (const float*)d_in[5];
  const float* Wv0 = (const float*)d_in[6];
  const float* bv0 = (const float*)d_in[7];
  const float* Wv1 = (const float*)d_in[8];
  const float* bv1 = (const float*)d_in[9];
  const float* Wv2 = (const float*)d_in[10];
  const float* bv2 = (const float*)d_in[11];
  const float* Wr  = (const float*)d_in[12];
  const float* br  = (const float*)d_in[13];
  float* out = (float*)d_out;

  char* ws = (char*)d_ws;
  size_t off = 0;
  auto take = [&](size_t bytes) -> void* {
    void* p = ws + off;
    off += (bytes + 255) & ~(size_t)255;
    return p;
  };
  __bf16* Wv0b  = (__bf16*)take((size_t)Hn * Sn * 2);        // 64 KB
  __bf16* Wv1b  = (__bf16*)take((size_t)Hn * Hn * 2);        // 128 KB
  __bf16* Wv2b  = (__bf16*)take((size_t)Dn * Sn * Hn * 2);   // 512 KB
  float*  lvl1w = (float*)take((size_t)Bn * Nn * Dn * 4);    // 1 MB
  float*  Cw    = (float*)take((size_t)Bn * Nn * 64 * 4);    // 8 MB
  float*  hid   = (float*)take((size_t)Bn * (Nn + 1) * Sn * 4);  // 16.9 MB

  cvt_bf16_kernel<<<(Hn * Sn + 255) / 256, 256, 0, stream>>>(Wv0, Wv0b, Hn * Sn);
  cvt_bf16_kernel<<<(Hn * Hn + 255) / 256, 256, 0, stream>>>(Wv1, Wv1b, Hn * Hn);
  cvt_bf16_kernel<<<(Dn * Sn * Hn + 255) / 256, 256, 0, stream>>>(Wv2, Wv2b,
                                                                  Dn * Sn * Hn);
  sig_kernel<<<(Bn * Nn + 255) / 256, 256, 0, stream>>>(x, lvl1w, Cw);
  init_kernel<<<Bn, Hn, 0, stream>>>(x, Wi0, bi0, Wi1, bi1, hid);

  for (int n = 0; n < Nn; ++n) {
    step_kernel<<<Bn / 16, 256, SM_TOT, stream>>>(hid, Wv0b, Wv1b, Wv2b, bv0,
                                                  bv1, bv2, lvl1w, Cw, n);
  }

  const int total = Bn * (Nn + 1) * OUTn;
  readout_kernel<<<(total + 255) / 256, 256, 0, stream>>>(hid, Wr, br, out);
}
